// EncoderModel_38508676776122
// MI455X (gfx1250) — matrix operand order repaired and checked by reading, not compiled, not run on hardware
//
#include <hip/hip_runtime.h>
#include <math.h>

#define S_   512
#define D_   512
#define H_   8
#define HD_  64
#define L_   6
#define B_   32
#define DFF_ 2048
#define M_   (B_ * S_)   // 16384 tokens

typedef __bf16 bf16_t;
typedef __attribute__((ext_vector_type(16))) __bf16 v16bf;
typedef __attribute__((ext_vector_type(8)))  __bf16 v8bf;
typedef __attribute__((ext_vector_type(8)))  float  v8f;

// ---- helpers -------------------------------------------------------------

__device__ __forceinline__ bf16_t f2bf(float f) {
  unsigned u = __builtin_bit_cast(unsigned, f);
  unsigned short s = (unsigned short)((u + 0x7FFFu + ((u >> 16) & 1u)) >> 16);
  return __builtin_bit_cast(bf16_t, s);
}

// A fragment: 16x32 bf16, A stored [M,K] row-major (global or LDS).
// Lane L: row = L&15; e=0..7 -> K = k0 + 8*(L>>4) + e ; e=8..15 -> +16.
__device__ __forceinline__ v16bf frag_a_g(const bf16_t* base, int ld) {
  const int lane = threadIdx.x & 31;
  const bf16_t* p = base + (size_t)(lane & 15) * ld + 8 * (lane >> 4);
  v8bf lo = *(const v8bf*)p;
  v8bf hi = *(const v8bf*)(p + 16);
  v16bf f;
#pragma unroll
  for (int i = 0; i < 8; ++i) { f[i] = lo[i]; f[i + 8] = hi[i]; }
  return f;
}

// B fragment: 32x16 bf16, B supplied transposed as Bt [N,K] row-major.
// Lane L: col = L&15; e=0..7 -> K = k0 + 8*(L>>4) + e ; e=8..15 -> +16.
__device__ __forceinline__ v16bf frag_b_g(const bf16_t* base, int ld) {
  const int lane = threadIdx.x & 31;
  const bf16_t* p = base + (size_t)(lane & 15) * ld + 8 * (lane >> 4);
  v8bf lo = *(const v8bf*)p;
  v8bf hi = *(const v8bf*)(p + 16);
  v16bf f;
#pragma unroll
  for (int i = 0; i < 8; ++i) { f[i] = lo[i]; f[i + 8] = hi[i]; }
  return f;
}

__device__ __forceinline__ v8f wmma_bf16(v16bf a, v16bf b, v8f c) {
  return __builtin_amdgcn_wmma_f32_16x16x32_bf16(false, a, false, b, (short)0, c,
                                                 false, false);
}

// 64x64 output tile per wave: 4x4 wmma accumulators, K-loop in steps of 32.
__device__ __forceinline__ void wave_gemm_64x64(const bf16_t* A, const bf16_t* Bt,
                                                int K, int m0, int n0,
                                                v8f acc[4][4]) {
#pragma unroll
  for (int i = 0; i < 4; ++i)
#pragma unroll
    for (int j = 0; j < 4; ++j)
#pragma unroll
      for (int e = 0; e < 8; ++e) acc[i][j][e] = 0.0f;

  for (int k0 = 0; k0 < K; k0 += 32) {
    v16bf a[4];
#pragma unroll
    for (int i = 0; i < 4; ++i)
      a[i] = frag_a_g(A + (size_t)(m0 + 16 * i) * K + k0, K);
#pragma unroll
    for (int j = 0; j < 4; ++j) {
      v16bf b = frag_b_g(Bt + (size_t)(n0 + 16 * j) * K + k0, K);
#pragma unroll
      for (int i = 0; i < 4; ++i) acc[i][j] = wmma_bf16(a[i], b, acc[i][j]);
    }
  }
}

// ---- weight prep ---------------------------------------------------------

// WqkvT[l][which*512 + h*64 + e][d] = W{q,k,v}[l][h][d][e]  (bf16)
__global__ __launch_bounds__(256) void conv_qkvT_kernel(
    const float* __restrict__ Wq, const float* __restrict__ Wk,
    const float* __restrict__ Wv, bf16_t* __restrict__ dst) {
  size_t idx = (size_t)blockIdx.x * 256 + threadIdx.x;  // L*1536*512
  int k = (int)(idx & 511);
  size_t t = idx >> 9;
  int n = (int)(t % 1536);
  int l = (int)(t / 1536);
  int which = n >> 9, rem = n & 511, h = rem >> 6, e = rem & 63;
  const float* W = (which == 0) ? Wq : (which == 1) ? Wk : Wv;
  dst[idx] = f2bf(W[(((size_t)l * H_ + h) * D_ + k) * HD_ + e]);
}

// dst[l][c][r] = src[l][r][c]  (bf16 transpose per layer)
__global__ __launch_bounds__(256) void conv_T_kernel(const float* __restrict__ src,
                                                     bf16_t* __restrict__ dst,
                                                     int R, int C) {
  size_t idx = (size_t)blockIdx.x * 256 + threadIdx.x;  // L*R*C
  int r = (int)(idx % R);
  size_t t = idx / R;
  int c = (int)(t % C);
  int l = (int)(t / C);
  dst[idx] = f2bf(src[((size_t)l * R + r) * C + c]);
}

// ---- embedding -----------------------------------------------------------

__global__ __launch_bounds__(256) void embed_kernel(const int* __restrict__ tokens,
                                                    const float* __restrict__ emb,
                                                    const float* __restrict__ pe,
                                                    float* __restrict__ xf,
                                                    bf16_t* __restrict__ xb) {
  size_t idx = (size_t)blockIdx.x * 256 + threadIdx.x;  // M_*D_
  int m = (int)(idx >> 9), c = (int)(idx & 511);
  int s = m & (S_ - 1);
  int tok = tokens[m];
  float v = emb[(size_t)tok * D_ + c] + pe[(size_t)s * D_ + c];
  xf[idx] = v;
  xb[idx] = f2bf(v);
}

// ---- GEMM kernels --------------------------------------------------------

// QKV projection: A [M,512] bf16, Bt [1536,512] bf16. Scatter q,k -> [B,H,S,HD],
// v -> transposed [B,H,HD,S] so the attention P*V B-fragments load contiguously.
__global__ __launch_bounds__(256) void gemm_qkv_kernel(
    const bf16_t* __restrict__ A, const bf16_t* __restrict__ Bt,
    bf16_t* __restrict__ qb, bf16_t* __restrict__ kb, bf16_t* __restrict__ vTb) {
  const int wid = threadIdx.x >> 5;
  const int m0 = blockIdx.y * 128 + (wid & 1) * 64;
  const int n0 = blockIdx.x * 256 + (wid >> 1) * 64;
  v8f acc[4][4];
  wave_gemm_64x64(A, Bt, D_, m0, n0, acc);
  const int lane = threadIdx.x & 31, half = lane >> 4, c15 = lane & 15;
#pragma unroll
  for (int i = 0; i < 4; ++i)
#pragma unroll
    for (int j = 0; j < 4; ++j)
#pragma unroll
      for (int r = 0; r < 8; ++r) {
        int row = m0 + 16 * i + r + 8 * half;
        int col = n0 + 16 * j + c15;
        int b = row >> 9, s = row & (S_ - 1);
        int which = col >> 9, rem = col & 511, h = rem >> 6, e = rem & 63;
        bf16_t v = f2bf(acc[i][j][r]);
        size_t bh = (size_t)(b * H_ + h);
        if (which == 0)      qb[(bh * S_ + s) * HD_ + e] = v;
        else if (which == 1) kb[(bh * S_ + s) * HD_ + e] = v;
        else                 vTb[(bh * HD_ + e) * S_ + s] = v;
      }
}

// C[M,N] f32 = A[M,K] @ Bt[N,K]^T (+ optional bias)
__global__ __launch_bounds__(256) void gemm_f32_kernel(
    const bf16_t* __restrict__ A, const bf16_t* __restrict__ Bt,
    const float* __restrict__ bias, float* __restrict__ C, int N, int K) {
  const int wid = threadIdx.x >> 5;
  const int m0 = blockIdx.y * 128 + (wid & 1) * 64;
  const int n0 = blockIdx.x * 256 + (wid >> 1) * 64;
  v8f acc[4][4];
  wave_gemm_64x64(A, Bt, K, m0, n0, acc);
  const int lane = threadIdx.x & 31, half = lane >> 4, c15 = lane & 15;
#pragma unroll
  for (int i = 0; i < 4; ++i)
#pragma unroll
    for (int j = 0; j < 4; ++j)
#pragma unroll
      for (int r = 0; r < 8; ++r) {
        int row = m0 + 16 * i + r + 8 * half;
        int col = n0 + 16 * j + c15;
        float v = acc[i][j][r];
        if (bias != nullptr) v += bias[col];
        C[(size_t)row * N + col] = v;
      }
}

// C[M,N] bf16 = gelu(A @ Bt^T + bias), exact erf gelu
__global__ __launch_bounds__(256) void gemm_gelu_kernel(
    const bf16_t* __restrict__ A, const bf16_t* __restrict__ Bt,
    const float* __restrict__ bias, bf16_t* __restrict__ C, int N, int K) {
  const int wid = threadIdx.x >> 5;
  const int m0 = blockIdx.y * 128 + (wid & 1) * 64;
  const int n0 = blockIdx.x * 256 + (wid >> 1) * 64;
  v8f acc[4][4];
  wave_gemm_64x64(A, Bt, K, m0, n0, acc);
  const int lane = threadIdx.x & 31, half = lane >> 4, c15 = lane & 15;
#pragma unroll
  for (int i = 0; i < 4; ++i)
#pragma unroll
    for (int j = 0; j < 4; ++j)
#pragma unroll
      for (int r = 0; r < 8; ++r) {
        int row = m0 + 16 * i + r + 8 * half;
        int col = n0 + 16 * j + c15;
        float v = acc[i][j][r] + bias[col];
        float g = 0.5f * v * (1.0f + erff(v * 0.70710678118654752f));
        C[(size_t)row * N + col] = f2bf(g);
      }
}

// ---- attention -----------------------------------------------------------
// One block per (b, h, 64-row query tile).
// LDS layout:
//   sc     : 64 x 520 f32 scores (phase 1/2); aliased as split-K obuf in phase 3
//   rowinv : 64 f32 softmax 1/sum
//   pb     : 64 x 520 bf16 unnormalized probabilities (written once in phase 2,
//            consumed by WMMA fragments via ds_load_b128 in phase 3)
__global__ __launch_bounds__(256) void attention_kernel(
    const bf16_t* __restrict__ qb, const bf16_t* __restrict__ kb,
    const bf16_t* __restrict__ vTb, const int* __restrict__ mask,
    bf16_t* __restrict__ attn_out) {
  const int SC_LD = 520, PB_LD = 520, OB_LD = 68;
  extern __shared__ float smem[];
  float*  sc     = smem;                              // 64*520 f32
  float*  rowinv = smem + 64 * SC_LD;                 // 64 f32
  bf16_t* pb     = (bf16_t*)(smem + 64 * SC_LD + 64); // 64*520 bf16 (16B aligned)
  float*  obuf   = sc;                                // reuse after softmax

  const int idx = blockIdx.x;
  const int qt = idx & 7;      // S_/64 tiles
  const int bh = idx >> 3;
  const int b = bh >> 3;       // H_ == 8
  const int h = bh & 7;
  const int m0 = qt * 64;

  const int wid = threadIdx.x >> 5;
  const int lane = threadIdx.x & 31, half = lane >> 4, c15 = lane & 15;

  const bf16_t* qbh = qb + (size_t)bh * S_ * HD_;
  const bf16_t* kbh = kb + (size_t)bh * S_ * HD_;
  const bf16_t* vbh = vTb + (size_t)bh * HD_ * S_;

  // Phase 1: scores[64, 512] = (Q_tile @ K^T) * scale, masked
  {
    const int n0 = wid * 64;
    v8f acc[4][4];
    wave_gemm_64x64(qbh + (size_t)m0 * HD_, kbh, HD_, 0, n0, acc);
#pragma unroll
    for (int i = 0; i < 4; ++i)
#pragma unroll
      for (int j = 0; j < 4; ++j)
#pragma unroll
        for (int r = 0; r < 8; ++r) {
          int lrow = 16 * i + r + 8 * half;
          int col = n0 + 16 * j + c15;  // key position t
          float v = acc[i][j][r] * 0.125f;
          if (mask[b * S_ + col] == 0) v = -3.4028235e38f;
          sc[lrow * SC_LD + col] = v;
        }
  }
  __syncthreads();

  // Phase 2: softmax per row; emit unnormalized exp as bf16 into pb, 1/sum cached
  for (int rr = 0; rr < 8; ++rr) {
    int row = wid * 8 + rr;
    const float* prow = sc + row * SC_LD;
    bf16_t* pbrow = pb + row * PB_LD;
    float mx = -3.4028235e38f;
    for (int c = lane; c < S_; c += 32) mx = fmaxf(mx, prow[c]);
#pragma unroll
    for (int off = 16; off; off >>= 1) mx = fmaxf(mx, __shfl_xor(mx, off, 32));
    float sum = 0.0f;
    for (int c = lane; c < S_; c += 32) {
      float e = __expf(prow[c] - mx);
      pbrow[c] = f2bf(e);
      sum += e;
    }
#pragma unroll
    for (int off = 16; off; off >>= 1) sum += __shfl_xor(sum, off, 32);
    if (lane == 0) rowinv[row] = 1.0f / sum;
  }
  __syncthreads();

  // Phase 3: O[64, 64] = P @ V ; waves split HD into 4 col tiles x 2 K-halves.
  // A fragments come straight from the bf16 pb buffer (ds_load_b128 pairs).
  {
    const int ct = wid & 3, kh = wid >> 2;
    v8f acc[4];
#pragma unroll
    for (int i = 0; i < 4; ++i)
#pragma unroll
      for (int e = 0; e < 8; ++e) acc[i][e] = 0.0f;

    const int kbeg = kh * 256;
    for (int k0 = kbeg; k0 < kbeg + 256; k0 += 32) {
      v16bf bfrag = frag_b_g(vbh + (size_t)(ct * 16) * S_ + k0, S_);
#pragma unroll
      for (int i = 0; i < 4; ++i) {
        v16bf afrag = frag_a_g(pb + (16 * i) * PB_LD + k0, PB_LD);
        acc[i] = wmma_bf16(afrag, bfrag, acc[i]);
      }
    }
    if (kh == 1) {
#pragma unroll
      for (int i = 0; i < 4; ++i)
#pragma unroll
        for (int r = 0; r < 8; ++r) {
          int lrow = 16 * i + r + 8 * half;
          obuf[lrow * OB_LD + ct * 16 + c15] = acc[i][r];
        }
    }
    __syncthreads();
    if (kh == 0) {
#pragma unroll
      for (int i = 0; i < 4; ++i)
#pragma unroll
        for (int r = 0; r < 8; ++r) {
          int lrow = 16 * i + r + 8 * half;
          float v = (acc[i][r] + obuf[lrow * OB_LD + ct * 16 + c15]) * rowinv[lrow];
          int s = m0 + lrow;
          attn_out[((size_t)(b * S_ + s)) * D_ + h * HD_ + ct * 16 + c15] = f2bf(v);
        }
    }
  }
}

// ---- residual + layernorm -----------------------------------------------

__global__ __launch_bounds__(256) void add_ln_kernel(float* __restrict__ x,
                                                     const float* __restrict__ delta,
                                                     const float* __restrict__ g,
                                                     const float* __restrict__ bta,
                                                     bf16_t* __restrict__ xbf) {
  const int row = blockIdx.x, t = threadIdx.x;
  __shared__ float red[256];
  size_t base = (size_t)row * D_;
  float v0 = x[base + t] + delta[base + t];
  float v1 = x[base + t + 256] + delta[base + t + 256];
  red[t] = v0 + v1;
  __syncthreads();
  for (int off = 128; off; off >>= 1) {
    if (t < off) red[t] += red[t + off];
    __syncthreads();
  }
  float mu = red[0] * (1.0f / D_);
  __syncthreads();
  float d0 = v0 - mu, d1 = v1 - mu;
  red[t] = d0 * d0 + d1 * d1;
  __syncthreads();
  for (int off = 128; off; off >>= 1) {
    if (t < off) red[t] += red[t + off];
    __syncthreads();
  }
  float inv = rsqrtf(red[0] * (1.0f / D_) + 1e-5f);
  float y0 = d0 * inv * g[t] + bta[t];
  float y1 = d1 * inv * g[t + 256] + bta[t + 256];
  x[base + t] = y0;
  x[base + t + 256] = y1;
  xbf[base + t] = f2bf(y0);
  xbf[base + t + 256] = f2bf(y1);
}

__global__ __launch_bounds__(256) void copy_out_kernel(const float* __restrict__ src,
                                                       float* __restrict__ dst) {
  size_t idx = (size_t)blockIdx.x * 256 + threadIdx.x;
  dst[idx] = src[idx];
}

// ---- host-side orchestration --------------------------------------------

extern "C" void kernel_launch(void* const* d_in, const int* in_sizes, int n_in,
                              void* d_out, int out_size, void* d_ws, size_t ws_size,
                              hipStream_t stream) {
  (void)in_sizes; (void)n_in; (void)out_size; (void)ws_size;
  const int*   tokens = (const int*)d_in[0];
  const int*   mask   = (const int*)d_in[1];
  const float* emb    = (const float*)d_in[2];
  const float* pe     = (const float*)d_in[3];
  const float* Wq     = (const float*)d_in[4];
  const float* Wk     = (const float*)d_in[5];
  const float* Wv     = (const float*)d_in[6];
  const float* Wo     = (const float*)d_in[7];
  const float* ln1g   = (const float*)d_in[8];
  const float* ln1b   = (const float*)d_in[9];
  const float* W1     = (const float*)d_in[10];
  const float* b1     = (const float*)d_in[11];
  const float* W2     = (const float*)d_in[12];
  const float* b2     = (const float*)d_in[13];
  const float* ln2g   = (const float*)d_in[14];
  const float* ln2b   = (const float*)d_in[15];

  char* ws = (char*)d_ws;
  size_t off = 0;
  float*  x_f32 = (float*)(ws + off);  off += (size_t)M_ * D_ * 4;
  bf16_t* x_bf  = (bf16_t*)(ws + off); off += (size_t)M_ * D_ * 2;
  bf16_t* qb    = (bf16_t*)(ws + off); off += (size_t)M_ * D_ * 2;
  bf16_t* kb    = (bf16_t*)(ws + off); off += (size_t)M_ * D_ * 2;
  bf16_t* vTb   = (bf16_t*)(ws + off); off += (size_t)M_ * D_ * 2;
  bf16_t* attn  = (bf16_t*)(ws + off); off += (size_t)M_ * D_ * 2;
  float*  tmpf  = (float*)(ws + off);  off += (size_t)M_ * D_ * 4;
  bf16_t* h1    = (bf16_t*)(ws + off); off += (size_t)M_ * DFF_ * 2;
  bf16_t* WqkvT = (bf16_t*)(ws + off); off += (size_t)L_ * 3 * D_ * D_ * 2;
  bf16_t* WoT   = (bf16_t*)(ws + off); off += (size_t)L_ * D_ * D_ * 2;
  bf16_t* W1T   = (bf16_t*)(ws + off); off += (size_t)L_ * D_ * DFF_ * 2;
  bf16_t* W2T   = (bf16_t*)(ws + off); off += (size_t)L_ * D_ * DFF_ * 2;

  // One-time (per call) weight conversion to transposed bf16
  conv_qkvT_kernel<<<(L_ * 3 * D_ * D_) / 256, 256, 0, stream>>>(Wq, Wk, Wv, WqkvT);
  conv_T_kernel<<<(L_ * D_ * D_) / 256, 256, 0, stream>>>(Wo, WoT, D_, D_);
  conv_T_kernel<<<(L_ * D_ * DFF_) / 256, 256, 0, stream>>>(W1, W1T, D_, DFF_);
  conv_T_kernel<<<(L_ * D_ * DFF_) / 256, 256, 0, stream>>>(W2, W2T, DFF_, D_);

  embed_kernel<<<(M_ * D_) / 256, 256, 0, stream>>>(tokens, emb, pe, x_f32, x_bf);

  // sc(64*520 f32) + rowinv(64 f32) + pb(64*520 bf16) = 199,936 B of 320K WGP LDS
  const int smem_attn = (64 * 520 + 64) * 4 + 64 * 520 * 2;
  for (int l = 0; l < L_; ++l) {
    gemm_qkv_kernel<<<dim3(6, 128), 256, 0, stream>>>(
        x_bf, WqkvT + (size_t)l * 3 * D_ * D_, qb, kb, vTb);
    attention_kernel<<<B_ * H_ * (S_ / 64), 256, smem_attn, stream>>>(
        qb, kb, vTb, mask, attn);
    gemm_f32_kernel<<<dim3(2, 128), 256, 0, stream>>>(
        attn, WoT + (size_t)l * D_ * D_, nullptr, tmpf, D_, D_);
    add_ln_kernel<<<M_, 256, 0, stream>>>(x_f32, tmpf, ln1g + l * D_, ln1b + l * D_, x_bf);
    gemm_gelu_kernel<<<dim3(8, 128), 256, 0, stream>>>(
        x_bf, W1T + (size_t)l * D_ * DFF_, b1 + l * DFF_, h1, DFF_, D_);
    gemm_f32_kernel<<<dim3(2, 128), 256, 0, stream>>>(
        h1, W2T + (size_t)l * D_ * DFF_, b2 + l * D_, tmpf, D_, DFF_);
    add_ln_kernel<<<M_, 256, 0, stream>>>(x_f32, tmpf, ln2g + l * D_, ln2b + l * D_, x_bf);
  }

  copy_out_kernel<<<(M_ * D_) / 256, 256, 0, stream>>>(x_f32, (float*)d_out);
}
